// ArlowVisionTransformer_24945170055308
// MI455X (gfx1250) — compile-verified, matrix operand order checked by reading, not run on hardware
//
#include <hip/hip_runtime.h>
#include <hip/hip_bf16.h>
#include <math.h>

// ---------------------------------------------------------------------------
// CDNA5 (gfx1250) ViT forward. wave32, v_wmma_f32_16x16x32_bf16 for all GEMMs.
// ---------------------------------------------------------------------------

typedef __attribute__((ext_vector_type(16))) __bf16 v16bf;
typedef __attribute__((ext_vector_type(8)))  float  v8f;

#define S_TOK   1024
#define D_EMB   1280
#define N_HEADS 16
#define HD      80
#define D_MLP   5120
#define D_OUT   3584

#define EP_BIAS      1
#define EP_RESID     2
#define EP_GELU_TANH 4
#define EP_GELU_ERF  8

__device__ __forceinline__ unsigned int f2bf(float f) {
  unsigned int u = __float_as_uint(f);
  u += 0x7FFFu + ((u >> 16) & 1u);
  return u >> 16;
}

// K offset of the even element held in dword j of a lane, within a 16x32
// bf16 A-fragment (per ISA 16-bit A-matrix 16x32 layout). B uses the mirrored
// layout (lane&15 -> N column, same K mapping).
__device__ __forceinline__ int kbase_of(int lane, int j) {
  return ((lane >> 4) & 1) * 8 + ((j >> 2) & 1) * 16 + ((j & 3) << 1);
}

__device__ __forceinline__ float gelu_tanh_f(float x) {
  float x3 = x * x * x;
  return 0.5f * x * (1.f + tanhf(0.7978845608028654f * (x + 0.044715f * x3)));
}
__device__ __forceinline__ float gelu_erf_f(float x) {
  return 0.5f * x * (1.f + erff(x * 0.7071067811865476f));
}

__device__ __forceinline__ v8f wmma_bf16(v16bf a, v16bf b, v8f c) {
  return __builtin_amdgcn_wmma_f32_16x16x32_bf16(false, a, false, b,
                                                 (short)0, c, false, false);
}

// ---------------------------------------------------------------------------
// Pack f32 row-major activation [M x K] -> bf16 A-fragment tiles.
// dst dword index == tid by construction: ((mt*Kt+kt)*32 + lane)*8 + j
// ---------------------------------------------------------------------------
__global__ void pack_a_kernel(const float* __restrict__ src,
                              unsigned int* __restrict__ dst,
                              int Mt, int Kt, int M, int K, int ld) {
  int tid = blockIdx.x * blockDim.x + threadIdx.x;
  int total = Mt * Kt * 256;
  if (tid >= total) return;
  int j = tid & 7, lane = (tid >> 3) & 31, tile = tid >> 8;
  int kt = tile % Kt, mt = tile / Kt;
  int m = mt * 16 + (lane & 15);
  int k = kt * 32 + kbase_of(lane, j);
  float lo = 0.f, hi = 0.f;
  if (m < M) {
    if (k < K)     lo = src[(size_t)m * ld + k];
    if (k + 1 < K) hi = src[(size_t)m * ld + k + 1];
  }
  dst[tid] = f2bf(lo) | (f2bf(hi) << 16);
}

// ---------------------------------------------------------------------------
// Pack f32 weight W[N x K] (row-major, used as B[k][n] = W[n][k]) into bf16
// B-fragment tiles: ((kt*Nt+nt)*32 + lane)*8 + j
// ---------------------------------------------------------------------------
__global__ void pack_w_kernel(const float* __restrict__ W,
                              unsigned int* __restrict__ dst,
                              int Kt, int Nt, int N, int K) {
  int tid = blockIdx.x * blockDim.x + threadIdx.x;
  int total = Kt * Nt * 256;
  if (tid >= total) return;
  int j = tid & 7, lane = (tid >> 3) & 31, tile = tid >> 8;
  int nt = tile % Nt, kt = tile / Nt;
  int n = nt * 16 + (lane & 15);
  int k = kt * 32 + kbase_of(lane, j);
  float lo = 0.f, hi = 0.f;
  if (n < N) {
    if (k < K)     lo = W[(size_t)n * K + k];
    if (k + 1 < K) hi = W[(size_t)n * K + k + 1];
  }
  dst[tid] = f2bf(lo) | (f2bf(hi) << 16);
}

// ---------------------------------------------------------------------------
// Patch-embed im2col, packed directly into A-fragments. Token s = h*32+w.
// Feature k flattens (c, frame, p, q) to match conv_w's [1280][1176] layout.
// Mt=64, Kt=37 (K=1176 padded to 1184).
// ---------------------------------------------------------------------------
__global__ void im2col_pack_kernel(const float* __restrict__ px,
                                   unsigned int* __restrict__ dst) {
  int tid = blockIdx.x * blockDim.x + threadIdx.x;
  int total = 64 * 37 * 256;
  if (tid >= total) return;
  int j = tid & 7, lane = (tid >> 3) & 31, tile = tid >> 8;
  int kt = tile % 37, mt = tile / 37;
  int s = mt * 16 + (lane & 15);
  int k = kt * 32 + kbase_of(lane, j);
  int hh = s >> 5, ww = s & 31;
  float v[2] = {0.f, 0.f};
#pragma unroll
  for (int e = 0; e < 2; ++e) {
    int kk = k + e;
    if (kk < 1176) {
      int q = kk % 14, p = (kk / 14) % 14, f = (kk / 196) % 2, c = kk / 392;
      v[e] = px[(((size_t)(c * 2 + f)) * 448 + (hh * 14 + p)) * 448 + (ww * 14 + q)];
    }
  }
  dst[tid] = f2bf(v[0]) | (f2bf(v[1]) << 16);
}

// ---------------------------------------------------------------------------
// LayerNorm over D=1280, one block per row, 256 threads x 5 elems.
// ---------------------------------------------------------------------------
__global__ void ln_kernel(const float* __restrict__ x, const float* __restrict__ w,
                          const float* __restrict__ b, float* __restrict__ y) {
  int row = blockIdx.x, t = threadIdx.x;
  const float* xr = x + (size_t)row * D_EMB;
  float* yr = y + (size_t)row * D_EMB;
  float v[5], s = 0.f, s2 = 0.f;
#pragma unroll
  for (int i = 0; i < 5; ++i) {
    float e = xr[t + 256 * i];
    v[i] = e; s += e; s2 += e * e;
  }
  __shared__ float r1[256], r2[256];
  r1[t] = s; r2[t] = s2; __syncthreads();
  for (int o = 128; o > 0; o >>= 1) {
    if (t < o) { r1[t] += r1[t + o]; r2[t] += r2[t + o]; }
    __syncthreads();
  }
  float mean = r1[0] * (1.f / 1280.f);
  float var  = r2[0] * (1.f / 1280.f) - mean * mean;
  float rs = rsqrtf(var + 1e-6f);
#pragma unroll
  for (int i = 0; i < 5; ++i) {
    int c = t + 256 * i;
    yr[c] = (v[i] - mean) * rs * w[c] + b[c];
  }
}

// ---------------------------------------------------------------------------
// RoPE cos/sin: emb[j] with j in [0,80): j40=j%40; j40<20 -> row pos, else col.
// inv freq = 10000^(-fi/20).
// ---------------------------------------------------------------------------
__device__ __forceinline__ void rope_cs(int tok, int jdim, float* c, float* sn) {
  int j40 = (jdim >= 40) ? jdim - 40 : jdim;
  float pos; int fi;
  if (j40 < 20) { pos = (float)(tok >> 5); fi = j40; }
  else          { pos = (float)(tok & 31); fi = j40 - 20; }
  float a = pos * __expf(-0.4605170185988091f * (float)fi);
  *c = __cosf(a); *sn = __sinf(a);
}

// Q with RoPE -> A-fragments per head: [1024 x 96pad], Mt=64, Kt=3.
__global__ void pack_q_rope_kernel(const float* __restrict__ qkv,
                                   unsigned int* __restrict__ dst) {
  int tid = blockIdx.x * blockDim.x + threadIdx.x;
  int total = N_HEADS * 192 * 256;
  if (tid >= total) return;
  int j = tid & 7, lane = (tid >> 3) & 31, tile = tid >> 8;
  int head = tile / 192, tl = tile % 192;
  int kt = tl % 3;
  int s = (tl / 3) * 16 + (lane & 15);
  int kk = kt * 32 + kbase_of(lane, j);
  float lo = 0.f, hi = 0.f;
  if (kk < HD) {
    const float* qr = qkv + (size_t)s * 3840 + head * HD;
    float q0 = qr[kk], q1 = qr[kk + 1];
    float c0, s0, c1, s1;
    rope_cs(s, kk, &c0, &s0); rope_cs(s, kk + 1, &c1, &s1);
    lo = q0 * c0 - q1 * s0;
    hi = q1 * c1 + q0 * s1;
  }
  dst[tid] = f2bf(lo) | (f2bf(hi) << 16);
}

// K with RoPE -> B-fragments per head: B[k=dim(96pad)][n=key token], Kt=3, Nt=64.
__global__ void pack_k_rope_kernel(const float* __restrict__ qkv,
                                   unsigned int* __restrict__ dst) {
  int tid = blockIdx.x * blockDim.x + threadIdx.x;
  int total = N_HEADS * 192 * 256;
  if (tid >= total) return;
  int j = tid & 7, lane = (tid >> 3) & 31, tile = tid >> 8;
  int head = tile / 192, tl = tile % 192;
  int nt = tl & 63, kt = tl >> 6;
  int tok = nt * 16 + (lane & 15);
  int kk = kt * 32 + kbase_of(lane, j);
  float lo = 0.f, hi = 0.f;
  if (kk < HD) {
    const float* kr = qkv + (size_t)tok * 3840 + 1280 + head * HD;
    float k0 = kr[kk], k1 = kr[kk + 1];
    float c0, s0, c1, s1;
    rope_cs(tok, kk, &c0, &s0); rope_cs(tok, kk + 1, &c1, &s1);
    lo = k0 * c0 - k1 * s0;
    hi = k1 * c1 + k0 * s1;
  }
  dst[tid] = f2bf(lo) | (f2bf(hi) << 16);
}

// V -> B-fragments per head: B[k=token(1024)][n=dim(80)], Kt=64, Nt=5.
__global__ void pack_v_kernel(const float* __restrict__ qkv,
                              unsigned int* __restrict__ dst) {
  int tid = blockIdx.x * blockDim.x + threadIdx.x;
  int total = N_HEADS * 320 * 256;
  if (tid >= total) return;
  int j = tid & 7, lane = (tid >> 3) & 31, tile = tid >> 8;
  int head = tile / 320, tl = tile % 320;
  int nt = tl % 5, kt = tl / 5;
  int d = nt * 16 + (lane & 15);
  int t0 = kt * 32 + kbase_of(lane, j);
  float lo = qkv[(size_t)t0 * 3840 + 2560 + head * HD + d];
  float hi = qkv[(size_t)(t0 + 1) * 3840 + 2560 + head * HD + d];
  dst[tid] = f2bf(lo) | (f2bf(hi) << 16);
}

// ---------------------------------------------------------------------------
// Softmax over 1024 keys (with 1/sqrt(80) scale), writes probabilities
// directly as bf16 A-fragments per head: [1024 x 1024], Mt=64, Kt=32.
// ---------------------------------------------------------------------------
__global__ void softmax_pack_kernel(const float* __restrict__ sc,
                                    unsigned int* __restrict__ pp) {
  int s = blockIdx.x, h = blockIdx.y, t = threadIdx.x;
  const float4* row = (const float4*)(sc + ((size_t)h * 1024 + s) * 1024);
  float4 q = row[t];
  const float scale = 0.11180339887498949f; // 80^-0.5
  float e0 = q.x * scale, e1 = q.y * scale, e2 = q.z * scale, e3 = q.w * scale;
  __shared__ float red[256];
  red[t] = fmaxf(fmaxf(e0, e1), fmaxf(e2, e3));
  __syncthreads();
  for (int o = 128; o > 0; o >>= 1) {
    if (t < o) red[t] = fmaxf(red[t], red[t + o]);
    __syncthreads();
  }
  float rmx = red[0];
  __syncthreads();
  e0 = __expf(e0 - rmx); e1 = __expf(e1 - rmx);
  e2 = __expf(e2 - rmx); e3 = __expf(e3 - rmx);
  red[t] = e0 + e1 + e2 + e3;
  __syncthreads();
  for (int o = 128; o > 0; o >>= 1) {
    if (t < o) red[t] += red[t + o];
    __syncthreads();
  }
  float inv = 1.f / red[0];
  e0 *= inv; e1 *= inv; e2 *= inv; e3 *= inv;
  int mt = s >> 4;
  size_t hbase = (size_t)h * 524288; // 2048 tiles * 256 dwords
  auto addr = [&](int k) -> size_t {
    int kt = k >> 5, kk = k & 31;
    int lhi = (kk >> 3) & 1;
    int jj = ((kk >> 4) & 1) * 4 + ((kk >> 1) & 3);
    int lane = lhi * 16 + (s & 15);
    return hbase + ((size_t)(mt * 32 + kt) * 32 + lane) * 8 + jj;
  };
  pp[addr(4 * t)]     = f2bf(e0) | (f2bf(e1) << 16);
  pp[addr(4 * t + 2)] = f2bf(e2) | (f2bf(e3) << 16);
}

// ---------------------------------------------------------------------------
// Batched WMMA GEMM: C[M,N] = A[M,K] * B[K,N] (+bias/resid/gelu).
// 8 waves per block; each wave owns a 32x64 output tile = 2 A-frags x
// 4 B-frags -> 8 WMMAs per 32-deep k-step, software-pipelined (next k-step's
// fragments are loaded before the current WMMAs issue). B-frags are shared
// between the two M-tiles, halving B traffic per WMMA.
// ---------------------------------------------------------------------------
__global__ void __launch_bounds__(256)
gemm_wmma_kernel(const unsigned short* __restrict__ Ap,
                 const unsigned short* __restrict__ Bp,
                 const float* __restrict__ bias,
                 const float* __restrict__ resid,
                 float* __restrict__ Cout,
                 int Kt, int Nt, int ldc, int Nvalid, int flags,
                 long long aBatch, long long bBatch, long long cBatch) {
  int lane = threadIdx.x & 31, wave = threadIdx.x >> 5;
  int bz = blockIdx.z;
  const v16bf* A = (const v16bf*)(Ap + (size_t)bz * aBatch);
  const v16bf* B = (const v16bf*)(Bp + (size_t)bz * bBatch);
  float* C = Cout + (size_t)bz * cBatch;
  int mt0 = (blockIdx.y * 8 + wave) * 2;
  int nt0 = blockIdx.x * 4;
  int ntv = Nt - nt0; if (ntv > 4) ntv = 4;

  v8f c00 = {}, c01 = {}, c02 = {}, c03 = {};
  v8f c10 = {}, c11 = {}, c12 = {}, c13 = {};

  const v16bf* pa0 = A + (size_t)mt0 * Kt * 32 + lane;
  const v16bf* pa1 = pa0 + (size_t)Kt * 32;
  const v16bf* pb  = B + (size_t)nt0 * 32 + lane;

  if (ntv == 4) {
    // fast path: full 4-wide N, branch-free double-buffered inner loop
    v16bf a0 = pa0[0];
    v16bf a1 = pa1[0];
    v16bf b0 = pb[0], b1 = pb[32], b2 = pb[64], b3 = pb[96];
    for (int kt = 0; kt < Kt; ++kt) {
      int kn = (kt + 1 < Kt) ? kt + 1 : kt;
      const v16bf* pbn = pb + (size_t)kn * Nt * 32;
      v16bf na0 = pa0[(size_t)kn * 32];
      v16bf na1 = pa1[(size_t)kn * 32];
      v16bf nb0 = pbn[0];
      v16bf nb1 = pbn[32];
      v16bf nb2 = pbn[64];
      v16bf nb3 = pbn[96];
      c00 = wmma_bf16(a0, b0, c00);
      c01 = wmma_bf16(a0, b1, c01);
      c02 = wmma_bf16(a0, b2, c02);
      c03 = wmma_bf16(a0, b3, c03);
      c10 = wmma_bf16(a1, b0, c10);
      c11 = wmma_bf16(a1, b1, c11);
      c12 = wmma_bf16(a1, b2, c12);
      c13 = wmma_bf16(a1, b3, c13);
      a0 = na0; a1 = na1;
      b0 = nb0; b1 = nb1; b2 = nb2; b3 = nb3;
    }
  } else {
    // ragged tail (only the last attn-out block, Nt=5)
    for (int kt = 0; kt < Kt; ++kt) {
      v16bf a0 = pa0[(size_t)kt * 32];
      v16bf a1 = pa1[(size_t)kt * 32];
      const v16bf* pbk = pb + (size_t)kt * Nt * 32;
      v16bf bb = pbk[0];
      c00 = wmma_bf16(a0, bb, c00);
      c10 = wmma_bf16(a1, bb, c10);
      if (ntv > 1) { bb = pbk[32]; c01 = wmma_bf16(a0, bb, c01); c11 = wmma_bf16(a1, bb, c11); }
      if (ntv > 2) { bb = pbk[64]; c02 = wmma_bf16(a0, bb, c02); c12 = wmma_bf16(a1, bb, c12); }
      if (ntv > 3) { bb = pbk[96]; c03 = wmma_bf16(a0, bb, c03); c13 = wmma_bf16(a1, bb, c13); }
    }
  }

  int rb = (lane & 16) ? 8 : 0;
  int col = lane & 15;
#define STORE_TILE(T, MT, ACC)                                                 \
  if ((T) < ntv) {                                                             \
    int n = (nt0 + (T)) * 16 + col;                                            \
    if (n < Nvalid) {                                                          \
      float bv = (flags & EP_BIAS) ? bias[n] : 0.f;                            \
      for (int r = 0; r < 8; ++r) {                                            \
        int m = (MT) * 16 + rb + r;                                            \
        float v = (ACC)[r] + bv;                                               \
        if (flags & EP_GELU_TANH) v = gelu_tanh_f(v);                          \
        if (flags & EP_GELU_ERF)  v = gelu_erf_f(v);                           \
        if (flags & EP_RESID)     v += resid[(size_t)m * ldc + n];             \
        C[(size_t)m * ldc + n] = v;                                            \
      }                                                                        \
    }                                                                          \
  }
  STORE_TILE(0, mt0, c00)
  STORE_TILE(1, mt0, c01)
  STORE_TILE(2, mt0, c02)
  STORE_TILE(3, mt0, c03)
  STORE_TILE(0, mt0 + 1, c10)
  STORE_TILE(1, mt0 + 1, c11)
  STORE_TILE(2, mt0 + 1, c12)
  STORE_TILE(3, mt0 + 1, c13)
#undef STORE_TILE
}

// ---------------------------------------------------------------------------
// Host orchestration.
// ---------------------------------------------------------------------------
extern "C" void kernel_launch(void* const* d_in, const int* in_sizes, int n_in,
                              void* d_out, int out_size, void* d_ws, size_t ws_size,
                              hipStream_t stream) {
  (void)in_sizes; (void)n_in; (void)out_size; (void)ws_size;
  const float* pixel  = (const float*)d_in[0];
  const float* conv_w = (const float*)d_in[1];
  const float* ln1_w  = (const float*)d_in[2];
  const float* ln1_b  = (const float*)d_in[3];
  const float* qkv_w  = (const float*)d_in[4];
  const float* qkv_b  = (const float*)d_in[5];
  const float* proj_w = (const float*)d_in[6];
  const float* proj_b = (const float*)d_in[7];
  const float* ln2_w  = (const float*)d_in[8];
  const float* ln2_b  = (const float*)d_in[9];
  const float* fc1_w  = (const float*)d_in[10];
  const float* fc1_b  = (const float*)d_in[11];
  const float* fc2_w  = (const float*)d_in[12];
  const float* fc2_b  = (const float*)d_in[13];
  const float* mln_w  = (const float*)d_in[14];
  const float* mln_b  = (const float*)d_in[15];
  const float* m0_w   = (const float*)d_in[16];
  const float* m0_b   = (const float*)d_in[17];
  const float* m2_w   = (const float*)d_in[18];
  const float* m2_b   = (const float*)d_in[19];

  char* ws = (char*)d_ws;
  size_t off = 0;
  auto carve = [&](size_t bytes) -> char* {
    char* p = ws + off;
    off += (bytes + 255) & ~(size_t)255;
    return p;
  };
  unsigned short* WPK = (unsigned short*)carve(56ull << 20);            // packed weights (max 50MB: m0)
  unsigned short* APK = (unsigned short*)carve(12ull << 20);            // packed activations (max 10MB)
  unsigned short* QP  = (unsigned short*)carve((size_t)16 * 192 * 1024);
  unsigned short* KP  = (unsigned short*)carve((size_t)16 * 192 * 1024);
  unsigned short* VP  = (unsigned short*)carve((size_t)16 * 320 * 1024);
  unsigned short* PP  = (unsigned short*)carve((size_t)16 * 2048 * 1024);
  float* SC  = (float*)carve((size_t)16 * 1024 * 1024 * 4);
  float* X0  = (float*)carve((size_t)1024 * 1280 * 4);
  float* X1  = (float*)carve((size_t)1024 * 1280 * 4);
  float* YLN = (float*)carve((size_t)1024 * 1280 * 4);
  float* QKV = (float*)carve((size_t)1024 * 3840 * 4);
  float* OAT = (float*)carve((size_t)1024 * 1280 * 4);
  float* HML = (float*)carve((size_t)1024 * 5120 * 4);
  float* H1  = (float*)carve((size_t)256 * 5120 * 4);

  auto packA = [&](const float* src, unsigned short* dst, int Mt, int Kt,
                   int M, int K, int ld) {
    int total = Mt * Kt * 256;
    pack_a_kernel<<<(total + 255) / 256, 256, 0, stream>>>(
        src, (unsigned int*)dst, Mt, Kt, M, K, ld);
  };
  auto packW = [&](const float* W, unsigned short* dst, int Kt, int Nt,
                   int N, int K) {
    int total = Kt * Nt * 256;
    pack_w_kernel<<<(total + 255) / 256, 256, 0, stream>>>(
        W, (unsigned int*)dst, Kt, Nt, N, K);
  };
  auto gemm = [&](const unsigned short* Apk, const unsigned short* Bpk,
                  const float* bias, const float* resid, float* C,
                  int Mt, int Kt, int Nt, int ldc, int Nvalid, int flags,
                  int batch, long long aB, long long bB, long long cB) {
    dim3 grid((Nt + 3) / 4, Mt / 16, batch);
    gemm_wmma_kernel<<<grid, 256, 0, stream>>>(
        Apk, Bpk, bias, resid, C, Kt, Nt, ldc, Nvalid, flags, aB, bB, cB);
  };

  // ---- Patch embedding: [1024 x 1176] x [1176 x 1280] ----
  packW(conv_w, WPK, 37, 80, 1280, 1176);
  {
    int total = 64 * 37 * 256;
    im2col_pack_kernel<<<(total + 255) / 256, 256, 0, stream>>>(
        pixel, (unsigned int*)APK);
  }
  gemm(APK, WPK, nullptr, nullptr, X0, 64, 37, 80, 1280, 1280, 0, 1, 0, 0, 0);

  float* cur = X0;
  float* oth = X1;

  for (int l = 0; l < 8; ++l) {
    // LN1 -> QKV
    ln_kernel<<<1024, 256, 0, stream>>>(cur, ln1_w + l * 1280, ln1_b + l * 1280, YLN);
    packA(YLN, APK, 64, 40, 1024, 1280, 1280);
    packW(qkv_w + (size_t)l * 3840 * 1280, WPK, 40, 240, 3840, 1280);
    gemm(APK, WPK, qkv_b + l * 3840, nullptr, QKV, 64, 40, 240, 3840, 3840,
         EP_BIAS, 1, 0, 0, 0);

    // RoPE + fragment packing of Q/K/V
    {
      int tq = N_HEADS * 192 * 256;
      pack_q_rope_kernel<<<(tq + 255) / 256, 256, 0, stream>>>(QKV, (unsigned int*)QP);
      pack_k_rope_kernel<<<(tq + 255) / 256, 256, 0, stream>>>(QKV, (unsigned int*)KP);
      int tv = N_HEADS * 320 * 256;
      pack_v_kernel<<<(tv + 255) / 256, 256, 0, stream>>>(QKV, (unsigned int*)VP);
    }

    // scores[h] = Q[h] x K[h]^T : M=1024, K=96, N=1024, batched over 16 heads
    gemm(QP, KP, nullptr, nullptr, SC, 64, 3, 64, 1024, 1024, 0, 16,
         (long long)192 * 512, (long long)192 * 512, (long long)1024 * 1024);

    softmax_pack_kernel<<<dim3(1024, 16), 256, 0, stream>>>(SC, (unsigned int*)PP);

    // O[h] = P[h] x V[h] : M=1024, K=1024, N=80; heads interleave into [1024,1280]
    gemm(PP, VP, nullptr, nullptr, OAT, 64, 32, 5, 1280, 80, 0, 16,
         (long long)2048 * 512, (long long)320 * 512, 80);

    // proj + residual
    packA(OAT, APK, 64, 40, 1024, 1280, 1280);
    packW(proj_w + (size_t)l * 1280 * 1280, WPK, 40, 80, 1280, 1280);
    gemm(APK, WPK, proj_b + l * 1280, cur, oth, 64, 40, 80, 1280, 1280,
         EP_BIAS | EP_RESID, 1, 0, 0, 0);

    // MLP
    ln_kernel<<<1024, 256, 0, stream>>>(oth, ln2_w + l * 1280, ln2_b + l * 1280, YLN);
    packA(YLN, APK, 64, 40, 1024, 1280, 1280);
    packW(fc1_w + (size_t)l * 5120 * 1280, WPK, 40, 320, 5120, 1280);
    gemm(APK, WPK, fc1_b + l * 5120, nullptr, HML, 64, 40, 320, 5120, 5120,
         EP_BIAS | EP_GELU_TANH, 1, 0, 0, 0);
    packA(HML, APK, 64, 160, 1024, 5120, 5120);
    packW(fc2_w + (size_t)l * 1280 * 5120, WPK, 160, 80, 1280, 5120);
    gemm(APK, WPK, fc2_b + l * 1280, oth, cur, 64, 160, 80, 1280, 1280,
         EP_BIAS | EP_RESID, 1, 0, 0, 0);
  }

  // ---- Head: LN, view [256,5120], m0 (erf-GELU), m2 ----
  ln_kernel<<<1024, 256, 0, stream>>>(cur, mln_w, mln_b, YLN);
  packA(YLN, APK, 16, 160, 256, 5120, 5120);
  packW(m0_w, WPK, 160, 320, 5120, 5120);
  gemm(APK, WPK, m0_b, nullptr, H1, 16, 160, 320, 5120, 5120,
       EP_BIAS | EP_GELU_ERF, 1, 0, 0, 0);
  packA(H1, APK, 16, 160, 256, 5120, 5120);
  packW(m2_w, WPK, 160, 224, 3584, 5120);
  gemm(APK, WPK, m2_b, nullptr, (float*)d_out, 16, 160, 224, 3584, 3584,
       EP_BIAS, 1, 0, 0, 0);
}
